// GRU_15564961480885
// MI455X (gfx1250) — compile-verified, matrix operand order checked by reading, not run on hardware
//
#include <hip/hip_runtime.h>
#include <math.h>

// ---------------------------------------------------------------------------
// Bidirectional 2-layer GRU for MI455X (gfx1250).
//   bf16 WMMA (fp32 accum) + fp32 gates; operands pre-converted to bf16;
//   async global->LDS staging (ASYNCcnt), software-pipelined in the proj
//   GEMM; LDS-resident recurrent weights in the scan.
// ---------------------------------------------------------------------------

typedef __attribute__((ext_vector_type(16))) __bf16 v16bf;
typedef __attribute__((ext_vector_type(8)))  __bf16 v8bf;
typedef __attribute__((ext_vector_type(4)))  __bf16 v4bf;
typedef __attribute__((ext_vector_type(8)))  float  v8f;
typedef __attribute__((ext_vector_type(4)))  int    v4i_t;

#define B_ 64
#define T_ 512
#define E_ 512
#define H_ 512
#define G_ (3 * H_)
#define KC 64            // K-chunk staged in LDS (proj kernel)

#define AS1 __attribute__((address_space(1)))
#define AS3 __attribute__((address_space(3)))

// asynccnt completes in-order: waiting <=N after issuing N new copies
// guarantees all previously-issued copies have landed.
#define WAIT_ASYNC(n) asm volatile("s_wait_asynccnt " #n ::: "memory")

// --- async global->LDS 16B copy (GLOBAL_LOAD_ASYNC_TO_LDS_B128) ------------
__device__ __forceinline__ void async_cp16(const __bf16* g, __bf16* l) {
  __builtin_amdgcn_global_load_async_to_lds_b128((AS1 v4i_t*)g, (AS3 v4i_t*)l,
                                                 0, 0);
}

// --- WMMA helpers ----------------------------------------------------------
__device__ __forceinline__ v8f wmma_bf16(v16bf a, v16bf b, v8f c) {
  return __builtin_amdgcn_wmma_f32_16x16x32_bf16(false, a, false, b,
                                                 (short)0, c, false, false);
}

// Fragment from LDS tile sm[rows][stride] (row-major bf16). Serves both A
// ([M][K]) and B ([N][K] = row-striped W^T, ISA 7.12.2).
__device__ __forceinline__ v16bf lds_frag(const __bf16* sm, int stride,
                                          int row16, int k0) {
  const int lane = threadIdx.x & 31;
  const int r = row16 + (lane & 15);
  const int ks = k0 + ((lane >> 4) ? 8 : 0);
  const v8bf lo = *(const v8bf*)(sm + r * stride + ks);
  const v8bf hi = *(const v8bf*)(sm + r * stride + ks + 16);
  v16bf f;
#pragma unroll
  for (int e = 0; e < 8; ++e) { f[e] = lo[e]; f[8 + e] = hi[e]; }
  return f;
}

// --- fp32 -> bf16 one-time conversion --------------------------------------
__global__ __launch_bounds__(256) void cvt_kernel(const float* __restrict__ s,
                                                  __bf16* __restrict__ d,
                                                  int n) {
  const int i = (blockIdx.x * 256 + threadIdx.x) * 4;
  if (i + 3 < n) {
    const float4 v = *(const float4*)(s + i);
    v4bf o;
    o[0] = (__bf16)v.x; o[1] = (__bf16)v.y;
    o[2] = (__bf16)v.z; o[3] = (__bf16)v.w;
    *(v4bf*)(d + i) = o;
  }
}

// --- Kernel 1: gx = X @ W^T + b (both directions) --------------------------
// WG: 128x128 C-block, 8 waves (2M x 4N), wave = 4x2 16x16 tiles.
// Double-buffered LDS K-chunks; next chunk's async copies overlap wmma.
__global__ __launch_bounds__(256) void gru_proj_kernel(
    const __bf16* __restrict__ Xb,   // [M][K] bf16
    const __bf16* __restrict__ Wb,   // [2][N][K] bf16
    const float* __restrict__ bias,  // [2][N]
    float* __restrict__ out,         // [2][M][N] f32
    int M, int N, int K) {
  __shared__ __bf16 smA[2][128 * KC];
  __shared__ __bf16 smB[2][128 * KC];
  const int dir = blockIdx.z;
  const int m0 = blockIdx.y * 128;
  const int n0 = blockIdx.x * 128;
  const __bf16* A = Xb + (size_t)m0 * K;
  const __bf16* Bw = Wb + (size_t)dir * N * K + (size_t)n0 * K;

  const int lane = threadIdx.x & 31;
  const int wave = threadIdx.x >> 5;
  const int wm = wave >> 2;  // 0..1 -> 64 rows
  const int wn = wave & 3;   // 0..3 -> 32 cols

  // 8 async instructions per stage() call (per wave).
  auto stage = [&](int kc, int buf) {
#pragma unroll
    for (int i = 0; i < 4; ++i) {
      const int id = threadIdx.x + i * 256;  // 0..1023
      const int row = id >> 3, c = (id & 7) * 8;
      async_cp16(A + (size_t)row * K + kc + c, smA[buf] + row * KC + c);
      async_cp16(Bw + (size_t)row * K + kc + c, smB[buf] + row * KC + c);
    }
  };

  v8f acc[4][2] = {};
  stage(0, 0);
  const int nch = K / KC;
  for (int i = 0; i < nch; ++i) {
    const int cur = i & 1;
    if (i + 1 < nch) {
      stage((i + 1) * KC, cur ^ 1);
      WAIT_ASYNC(0x8);  // wait only for chunk i; chunk i+1 stays in flight
    } else {
      WAIT_ASYNC(0x0);
    }
    __syncthreads();
#pragma unroll
    for (int k0 = 0; k0 < KC; k0 += 32) {
      const v16bf bf0 = lds_frag(smB[cur], KC, wn * 32, k0);
      const v16bf bf1 = lds_frag(smB[cur], KC, wn * 32 + 16, k0);
#pragma unroll
      for (int mt = 0; mt < 4; ++mt) {
        const v16bf af = lds_frag(smA[cur], KC, wm * 64 + mt * 16, k0);
        acc[mt][0] = wmma_bf16(af, bf0, acc[mt][0]);
        acc[mt][1] = wmma_bf16(af, bf1, acc[mt][1]);
      }
    }
    __syncthreads();
  }

  const int hi = lane >> 4;
  const int ncol = lane & 15;
  float* od = out + (size_t)dir * M * N;
#pragma unroll
  for (int nt = 0; nt < 2; ++nt) {
    const int n = n0 + wn * 32 + nt * 16 + ncol;
    const float bv = bias[dir * N + n];
#pragma unroll
    for (int mt = 0; mt < 4; ++mt) {
      float* p = od + (size_t)(m0 + wm * 64 + mt * 16 + hi * 8) * N + n;
#pragma unroll
      for (int r = 0; r < 8; ++r) p[(size_t)r * N] = acc[mt][nt][r] + bv;
    }
  }
}

// --- device-scope split barrier (monotonic counter) ------------------------
__device__ __forceinline__ void grid_barrier(unsigned* cnt, unsigned target) {
  __threadfence();
  __syncthreads();
  if (threadIdx.x == 0) {
    atomicAdd(cnt, 1u);
    while (__hip_atomic_load(cnt, __ATOMIC_ACQUIRE, __HIP_MEMORY_SCOPE_AGENT) <
           target)
      __builtin_amdgcn_s_sleep(1);
  }
  __syncthreads();
}

// --- Kernel 2: persistent recurrent scan -----------------------------------
// 48 WGs = 24 per direction, each owns 64 cols of G.
//   W_hh slice (64x512 bf16 = 64KB) is LDS-resident for the whole scan.
//   Per step: async-stage h_prev (64KB bf16), one uninterrupted K=512 wmma
//   loop, gh -> global, grid barrier, gates on ALL 24 WGs, grid barrier.
__global__ __launch_bounds__(256) void gru_scan_kernel(
    const float* __restrict__ gx,     // [2][B*T][G] f32
    const __bf16* __restrict__ Whhb,  // [2][G][H] bf16
    const float* __restrict__ bhh,    // [2][G]
    float* __restrict__ hbuf,         // [2dir][2slot][B][H] f32 (zeroed)
    __bf16* __restrict__ hbufb,       // [2dir][2slot][B][H] bf16 (zeroed)
    float* __restrict__ ghbuf,        // [2dir][B][G]
    unsigned* __restrict__ sync,      // [2dir] (zeroed)
    __bf16* __restrict__ outb,        // [B][T][2H] bf16 (layer0) or null
    float* __restrict__ outf,         // [B][T][2H] f32 (layer1) or null
    float* __restrict__ out_hy,       // [4][B][H]
    int layer) {
  __shared__ __bf16 smW[64 * H_];  // 64KB resident W slice rows n0..n0+63
  __shared__ __bf16 smH[64 * H_];  // 64KB h_prev staged per step
  const int NWG = 24;
  const int dir = blockIdx.x / NWG;
  const int wg = blockIdx.x % NWG;
  const int n0 = wg * 64;  // slice of G

  const __bf16* Wd = Whhb + (size_t)dir * G_ * H_ + (size_t)n0 * H_;
  const float* gxd = gx + (size_t)dir * B_ * T_ * G_;
  const float* bd = bhh + (size_t)dir * G_;
  float* gh = ghbuf + (size_t)dir * B_ * G_;
  float* hb = hbuf + (size_t)dir * 2 * B_ * H_;
  __bf16* hbb = hbufb + (size_t)dir * 2 * B_ * H_;
  unsigned* cnt = sync + dir;
  unsigned target = 0;

  const int lane = threadIdx.x & 31;
  const int wave = threadIdx.x >> 5;
  const int mt = wave >> 1;  // 0..3 : batch-row tile
  const int nh = wave & 1;   // 0..1 : 32-col half

  // One-time preload of the W_hh slice (16 async instrs; wait folds into
  // the first step's WAIT_ASYNC(0)).
#pragma unroll
  for (int i = 0; i < 16; ++i) {
    const int c = threadIdx.x + i * 256;  // 4096 16B chunks
    const int row = c >> 6, off = (c & 63) * 8;
    async_cp16(Wd + (size_t)row * H_ + off, smW + row * H_ + off);
  }

  for (int t = 0; t < T_; ++t) {
    const int t_eff = dir ? (T_ - 1 - t) : t;
    const float* hprev = hb + (size_t)(t & 1) * B_ * H_;
    const __bf16* hprevb = hbb + (size_t)(t & 1) * B_ * H_;
    float* hnext = hb + (size_t)((t + 1) & 1) * B_ * H_;
    __bf16* hnextb = hbb + (size_t)((t + 1) & 1) * B_ * H_;

    // Stage h_prev (bf16) into LDS.
#pragma unroll
    for (int i = 0; i < 16; ++i) {
      const int c = threadIdx.x + i * 256;
      const int row = c >> 6, off = (c & 63) * 8;
      async_cp16(hprevb + (size_t)row * H_ + off, smH + row * H_ + off);
    }
    WAIT_ASYNC(0x0);
    __syncthreads();

    // Phase 1: gh[0:64, n0:n0+64] = h_prev @ Whh^T + b  (32 wmma/wave).
    v8f acc[2] = {};
#pragma unroll 4
    for (int k0 = 0; k0 < H_; k0 += 32) {
      const v16bf af = lds_frag(smH, H_, mt * 16, k0);
      const v16bf bf0 = lds_frag(smW, H_, nh * 32, k0);
      const v16bf bf1 = lds_frag(smW, H_, nh * 32 + 16, k0);
      acc[0] = wmma_bf16(af, bf0, acc[0]);
      acc[1] = wmma_bf16(af, bf1, acc[1]);
    }
    const int hi = lane >> 4;
    const int ncol = lane & 15;
#pragma unroll
    for (int nt2 = 0; nt2 < 2; ++nt2) {
      const int n = n0 + nh * 32 + nt2 * 16 + ncol;
      const float bv = bd[n];
      float* pg = gh + (size_t)(mt * 16 + hi * 8) * G_ + n;
#pragma unroll
      for (int r = 0; r < 8; ++r) pg[(size_t)r * G_] = acc[nt2][r] + bv;
    }
    target += NWG;
    grid_barrier(cnt, target);

    // Phase 2: fp32 gate math over all 64x512 elements, spread across the
    // direction's 24 WGs (gh is global; any WG can serve any element).
    {
      const int tid24 = wg * 256 + threadIdx.x;  // 0..6143
      for (int e = tid24; e < B_ * H_; e += NWG * 256) {
        const int b = e >> 9;        // /H_
        const int j = e & (H_ - 1);  // %H_
        const float* gxt = gxd + ((size_t)b * T_ + t_eff) * G_;
        const float* ghb = gh + (size_t)b * G_;
        const float r_ = 1.f / (1.f + __expf(-(gxt[j] + ghb[j])));
        const float z_ = 1.f / (1.f + __expf(-(gxt[H_ + j] + ghb[H_ + j])));
        const float n_ = tanhf(gxt[2 * H_ + j] + r_ * ghb[2 * H_ + j]);
        const float hN = (1.f - z_) * n_ + z_ * hprev[(size_t)b * H_ + j];
        hnext[(size_t)b * H_ + j] = hN;
        hnextb[(size_t)b * H_ + j] = (__bf16)hN;
        const size_t so =
            ((size_t)b * T_ + t_eff) * (2 * H_) + (size_t)dir * H_ + j;
        if (outb) outb[so] = (__bf16)hN;
        if (outf) outf[so] = hN;
        if (t == T_ - 1)
          out_hy[((size_t)(layer * 2 + dir) * B_ + b) * H_ + j] = hN;
      }
    }
    target += NWG;
    grid_barrier(cnt, target);  // also guards smH overwrite next step
  }
}

// ---------------------------------------------------------------------------

extern "C" void kernel_launch(void* const* d_in, const int* in_sizes, int n_in,
                              void* d_out, int out_size, void* d_ws,
                              size_t ws_size, hipStream_t stream) {
  (void)in_sizes; (void)n_in; (void)out_size; (void)ws_size;

  const float* xs    = (const float*)d_in[0];
  const float* w_ih0 = (const float*)d_in[1];
  const float* w_hh0 = (const float*)d_in[2];
  const float* b_ih0 = (const float*)d_in[3];
  const float* b_hh0 = (const float*)d_in[4];
  const float* w_ih1 = (const float*)d_in[5];
  const float* w_hh1 = (const float*)d_in[6];
  const float* b_ih1 = (const float*)d_in[7];
  const float* b_hh1 = (const float*)d_in[8];

  const size_t M = (size_t)B_ * T_;

  // Workspace layout (bytes).
  char* w = (char*)d_ws;
  float*  gx    = (float*)w;  w += 2ull * M * G_ * 4;        // 402.6 MB
  __bf16* xsb   = (__bf16*)w; w += (size_t)B_ * T_ * E_ * 2; // 33.5 MB
  __bf16* y0b   = (__bf16*)w; w += M * 2 * H_ * 2;           // 67.1 MB
  __bf16* wih0b = (__bf16*)w; w += 2ull * G_ * E_ * 2;
  __bf16* whh0b = (__bf16*)w; w += 2ull * G_ * H_ * 2;
  __bf16* wih1b = (__bf16*)w; w += 2ull * G_ * 2 * H_ * 2;
  __bf16* whh1b = (__bf16*)w; w += 2ull * G_ * H_ * 2;
  float*  hbuf  = (float*)w;  w += 2ull * 2 * 2 * B_ * H_ * 4;
  __bf16* hbufb = (__bf16*)w; w += 2ull * 2 * 2 * B_ * H_ * 2;
  float*  ghbuf = (float*)w;  w += 2ull * B_ * G_ * 4;
  unsigned* sync = (unsigned*)w;

  const size_t clear_bytes = 2ull * 2 * 2 * B_ * H_ * 4 +
                             2ull * 2 * 2 * B_ * H_ * 2 +
                             2ull * B_ * G_ * 4 + 4 * sizeof(unsigned);
  (void)hipMemsetAsync(hbuf, 0, clear_bytes, stream);  // h0 = 0, counters = 0

  float* out = (float*)d_out;
  float* hy  = out + M * 2 * H_;

  dim3 blk(256);
  // One-time bf16 conversion of xs + all weights.
  auto nblk = [](size_t n) { return (unsigned)(n / 1024); };
  cvt_kernel<<<nblk((size_t)B_ * T_ * E_), blk, 0, stream>>>(
      xs, xsb, B_ * T_ * E_);
  cvt_kernel<<<nblk(2ull * G_ * E_), blk, 0, stream>>>(w_ih0, wih0b,
                                                       2 * G_ * E_);
  cvt_kernel<<<nblk(2ull * G_ * H_), blk, 0, stream>>>(w_hh0, whh0b,
                                                       2 * G_ * H_);
  cvt_kernel<<<nblk(2ull * G_ * 2 * H_), blk, 0, stream>>>(w_ih1, wih1b,
                                                           2 * G_ * 2 * H_);
  cvt_kernel<<<nblk(2ull * G_ * H_), blk, 0, stream>>>(w_hh1, whh1b,
                                                       2 * G_ * H_);

  const size_t hlayer_f = 2ull * 2 * B_ * H_;  // per-layer h strides
  dim3 pgrid(G_ / 128, (unsigned)(M / 128), 2);

  // Layer 0
  gru_proj_kernel<<<pgrid, blk, 0, stream>>>(xsb, wih0b, b_ih0, gx, (int)M,
                                             G_, E_);
  gru_scan_kernel<<<48, blk, 0, stream>>>(gx, whh0b, b_hh0, hbuf, hbufb,
                                          ghbuf, sync, y0b, (float*)nullptr,
                                          hy, 0);
  // Layer 1 (input = y0 bf16, K = 2H)
  gru_proj_kernel<<<pgrid, blk, 0, stream>>>(y0b, wih1b, b_ih1, gx, (int)M,
                                             G_, 2 * H_);
  gru_scan_kernel<<<48, blk, 0, stream>>>(gx, whh1b, b_hh1, hbuf + hlayer_f,
                                          hbufb + hlayer_f, ghbuf, sync + 2,
                                          (__bf16*)nullptr, out, hy, 1);
}
